// LSTM_61186104099678
// MI455X (gfx1250) — matrix operand order repaired and checked by reading, not compiled, not run on hardware
//
#include <hip/hip_runtime.h>
#include <hip/hip_bf16.h>

typedef __attribute__((ext_vector_type(16))) __bf16 v16bf;
typedef __attribute__((ext_vector_type(8)))  float  v8f;
typedef unsigned short u16;
typedef unsigned int   u32;

#define B_   64
#define T_   512
#define E_   512
#define H_   1024
#define NG_  4096          // 4*H
#define K0_  1536          // E + H
#define K1_  2048          // 2*H
#define NBLK 64
#define NTHR 512           // 16 wave32 waves per block -> 1024 waves total

// ---------- bf16 packing helper (round-to-nearest-even) ----------
__device__ __forceinline__ u16 bf1(float x) {
    u32 u = __float_as_uint(x);
    return (u16)((u + 0x7FFFu + ((u >> 16) & 1u)) >> 16);
}

union Frag { uint4 u[2]; v16bf v; };   // 16 bf16 = 32 bytes = A or B operand per lane

__device__ __forceinline__ float sigf(float x) { return 1.0f / (1.0f + __expf(-x)); }

// ---------- grid-wide sense barrier ----------
__device__ __forceinline__ void grid_sync(int* cnt, volatile int* gen) {
    __syncthreads();
    if (threadIdx.x == 0) {
        __threadfence();
        int g = *gen;
        if (atomicAdd(cnt, 1) == NBLK - 1) {
            *cnt = 0;
            __threadfence();
            atomicAdd((int*)gen, 1);
        } else {
            while (*gen == g) { __builtin_amdgcn_s_sleep(1); }
        }
    }
    __syncthreads();
}

// ---------- prologue: zero state + barrier ----------
__global__ void init_state(int* bar, float* C0, float* C1, u16* H0, u16* H1, float* H1f) {
    int tid = blockIdx.x * blockDim.x + threadIdx.x;
    if (tid < 2) bar[tid] = 0;
    for (int i = tid; i < B_ * H_; i += gridDim.x * blockDim.x) {
        C0[i] = 0.0f; C1[i] = 0.0f; H1f[i] = 0.0f;
        H0[i] = 0;    H1[i] = 0;
    }
}

// ---------- prologue: fp32 -> bf16 weight conversion ----------
__global__ void convert_weights(const float* W0, const float* W1, u16* W0b, u16* W1b) {
    int stride = gridDim.x * blockDim.x;
    int tid = blockIdx.x * blockDim.x + threadIdx.x;
    for (int i = tid; i < NG_ * K0_; i += stride) W0b[i] = bf1(W0[i]);
    for (int i = tid; i < NG_ * K1_; i += stride) W1b[i] = bf1(W1[i]);
}

// ---------- prologue: gather + convert embeddings for the whole sequence ----------
// Xbf[(t*B + b)*E + k] = bf16( emb[x[b,t], k] )
__global__ void embed_gather(const int* __restrict__ x, const float* __restrict__ emb,
                             u16* __restrict__ Xbf) {
    int stride = gridDim.x * blockDim.x;
    int tid = blockIdx.x * blockDim.x + threadIdx.x;
    for (long i = tid; i < (long)T_ * B_ * E_; i += stride) {
        int k = (int)(i & (E_ - 1));
        long row = i >> 9;                 // t*B + b
        int b = (int)(row & (B_ - 1));
        int t = (int)(row >> 6);
        int tok = x[b * T_ + t];
        Xbf[i] = bf1(emb[(long)tok * E_ + k]);
    }
}

// ---------- persistent 2-layer LSTM ----------
__global__ __launch_bounds__(NTHR) void lstm_persistent(
    const u16* __restrict__ Xbf,
    const u16* __restrict__ W0b, const u16* __restrict__ W1b,
    const float* __restrict__ b0, const float* __restrict__ b1,
    float* __restrict__ G, float* __restrict__ C0, float* __restrict__ C1,
    u16* __restrict__ H0, u16* __restrict__ H1, float* __restrict__ H1f,
    int* bar)
{
    const int lane  = threadIdx.x & 31;
    const int wid   = (blockIdx.x * NTHR + threadIdx.x) >> 5;   // 0..1023
    const int mtile = wid & 3;                                  // 4 M-tiles (batch)
    const int ntile = wid >> 2;                                 // 256 N-tiles (gates)
    const int kh    = lane >> 4;                                // half-wave select
    const int lm    = lane & 15;
    const int mrow  = mtile * 16 + lm;                          // batch row for A loads
    const int nrow  = ntile * 16 + lm;                          // weight row for B loads
    const int gtid  = blockIdx.x * NTHR + threadIdx.x;          // elementwise id

    int* cnt = bar;
    volatile int* gen = bar + 1;

    const u16* h0r = H0 + mrow * H_;
    const u16* h1r = H1 + mrow * H_;
    const u16* w0r = W0b + (long)nrow * K0_;
    const u16* w1r = W1b + (long)nrow * K1_;

    for (int t = 0; t < T_; ++t) {
        // ---------------- Phase 1: gates0 = [x_t, h0] @ W0^T ----------------
        {
            v8f acc = {};
            const u16* xrow = Xbf + ((long)t * B_ + mrow) * E_;

            // K over the (pre-converted bf16) embedding part: 16 k-tiles
            #pragma clang loop unroll_count(4)
            for (int kt = 0; kt < E_ / 32; ++kt) {
                const int kb = kt * 32 + 8 * kh;
                Frag a;
                a.u[0] = *(const uint4*)(xrow + kb);
                a.u[1] = *(const uint4*)(xrow + kb + 16);
                const uint4* wp = (const uint4*)(w0r + kt * 32 + 8 * kh);
                __builtin_prefetch((const void*)(wp + 4), 0, 3);
                Frag b; b.u[0] = wp[0]; b.u[1] = wp[2];
                acc = __builtin_amdgcn_wmma_f32_16x16x32_bf16(
                          false, a.v, false, b.v, (short)0, acc, false, false);
            }
            // K over the recurrent part: 32 k-tiles
            #pragma clang loop unroll_count(4)
            for (int kt = 0; kt < H_ / 32; ++kt) {
                const int kb = kt * 32 + 8 * kh;
                Frag a;
                a.u[0] = *(const uint4*)(h0r + kb);
                a.u[1] = *(const uint4*)(h0r + kb + 16);
                const uint4* wp = (const uint4*)(w0r + E_ + kt * 32 + 8 * kh);
                __builtin_prefetch((const void*)(wp + 4), 0, 3);
                Frag b; b.u[0] = wp[0]; b.u[1] = wp[2];
                acc = __builtin_amdgcn_wmma_f32_16x16x32_bf16(
                          false, a.v, false, b.v, (short)0, acc, false, false);
            }
            #pragma unroll
            for (int v = 0; v < 8; ++v)
                G[(mtile * 16 + v + 8 * kh) * NG_ + ntile * 16 + lm] = acc[v];
        }
        grid_sync(cnt, gen);

        // ---------------- Phase 2: layer-0 cell update ----------------
        for (int e = gtid; e < B_ * H_; e += NBLK * NTHR) {
            const int b = e >> 10, j = e & (H_ - 1);
            const float* g = G + b * NG_;
            float ig = sigf (g[j]            + b0[j]);
            float fg = sigf (g[H_ + j]       + b0[H_ + j]);
            float gg = tanhf(g[2 * H_ + j]   + b0[2 * H_ + j]);
            float og = sigf (g[3 * H_ + j]   + b0[3 * H_ + j]);
            float c  = fg * C0[e] + ig * gg;
            C0[e] = c;
            H0[e] = bf1(og * tanhf(c));
        }
        grid_sync(cnt, gen);

        // ---------------- Phase 3: gates1 = [h0, h1] @ W1^T ----------------
        {
            v8f acc = {};
            #pragma clang loop unroll_count(4)
            for (int kt = 0; kt < H_ / 32; ++kt) {
                const int kb = kt * 32 + 8 * kh;
                Frag a;
                a.u[0] = *(const uint4*)(h0r + kb);
                a.u[1] = *(const uint4*)(h0r + kb + 16);
                const uint4* wp = (const uint4*)(w1r + kt * 32 + 8 * kh);
                __builtin_prefetch((const void*)(wp + 4), 0, 3);
                Frag b; b.u[0] = wp[0]; b.u[1] = wp[2];
                acc = __builtin_amdgcn_wmma_f32_16x16x32_bf16(
                          false, a.v, false, b.v, (short)0, acc, false, false);
            }
            #pragma clang loop unroll_count(4)
            for (int kt = 0; kt < H_ / 32; ++kt) {
                const int kb = kt * 32 + 8 * kh;
                Frag a;
                a.u[0] = *(const uint4*)(h1r + kb);
                a.u[1] = *(const uint4*)(h1r + kb + 16);
                const uint4* wp = (const uint4*)(w1r + H_ + kt * 32 + 8 * kh);
                __builtin_prefetch((const void*)(wp + 4), 0, 3);
                Frag b; b.u[0] = wp[0]; b.u[1] = wp[2];
                acc = __builtin_amdgcn_wmma_f32_16x16x32_bf16(
                          false, a.v, false, b.v, (short)0, acc, false, false);
            }
            #pragma unroll
            for (int v = 0; v < 8; ++v)
                G[(mtile * 16 + v + 8 * kh) * NG_ + ntile * 16 + lm] = acc[v];
        }
        grid_sync(cnt, gen);

        // ---------------- Phase 4: layer-1 cell update ----------------
        for (int e = gtid; e < B_ * H_; e += NBLK * NTHR) {
            const int b = e >> 10, j = e & (H_ - 1);
            const float* g = G + b * NG_;
            float ig = sigf (g[j]            + b1[j]);
            float fg = sigf (g[H_ + j]       + b1[H_ + j]);
            float gg = tanhf(g[2 * H_ + j]   + b1[2 * H_ + j]);
            float og = sigf (g[3 * H_ + j]   + b1[3 * H_ + j]);
            float c  = fg * C1[e] + ig * gg;
            C1[e] = c;
            float h = og * tanhf(c);
            H1[e]  = bf1(h);
            H1f[e] = h;
        }
        grid_sync(cnt, gen);
    }
}

// ---------- epilogue: out = h1_final @ V^T + bV ----------
__global__ void classifier(const float* __restrict__ H1f, const float* __restrict__ V,
                           const float* __restrict__ bV, float* __restrict__ out) {
    __shared__ float red[256];
    const int b = blockIdx.x;
    for (int c = 0; c < 10; ++c) {
        float s = 0.0f;
        for (int j = threadIdx.x; j < H_; j += 256)
            s += H1f[b * H_ + j] * V[c * H_ + j];
        red[threadIdx.x] = s; __syncthreads();
        for (int off = 128; off > 0; off >>= 1) {
            if (threadIdx.x < off) red[threadIdx.x] += red[threadIdx.x + off];
            __syncthreads();
        }
        if (threadIdx.x == 0) out[b * 10 + c] = red[0] + bV[c];
        __syncthreads();
    }
}

extern "C" void kernel_launch(void* const* d_in, const int* in_sizes, int n_in,
                              void* d_out, int out_size, void* d_ws, size_t ws_size,
                              hipStream_t stream) {
    const int*   x   = (const int*)  d_in[0];
    const float* emb = (const float*)d_in[1];
    const float* W0  = (const float*)d_in[2];
    const float* b0  = (const float*)d_in[3];
    const float* W1  = (const float*)d_in[4];
    const float* b1  = (const float*)d_in[5];
    const float* V   = (const float*)d_in[6];
    const float* bV  = (const float*)d_in[7];
    float* out = (float*)d_out;

    size_t off = 0;
    auto alloc = [&](size_t bytes) {
        void* p = (char*)d_ws + off;
        off += (bytes + 255) & ~(size_t)255;
        return p;
    };
    int*   bar  = (int*)  alloc(256);
    u16*   W0b  = (u16*)  alloc((size_t)NG_ * K0_ * 2);
    u16*   W1b  = (u16*)  alloc((size_t)NG_ * K1_ * 2);
    u16*   Xbf  = (u16*)  alloc((size_t)T_ * B_ * E_ * 2);
    float* G    = (float*)alloc((size_t)B_ * NG_ * 4);
    float* C0   = (float*)alloc((size_t)B_ * H_ * 4);
    float* C1   = (float*)alloc((size_t)B_ * H_ * 4);
    u16*   H0   = (u16*)  alloc((size_t)B_ * H_ * 2);
    u16*   H1   = (u16*)  alloc((size_t)B_ * H_ * 2);
    float* H1f  = (float*)alloc((size_t)B_ * H_ * 4);
    (void)ws_size; (void)in_sizes; (void)n_in; (void)out_size;

    init_state<<<64, 256, 0, stream>>>(bar, C0, C1, H0, H1, H1f);
    convert_weights<<<1024, 256, 0, stream>>>(W0, W1, W0b, W1b);
    embed_gather<<<1024, 256, 0, stream>>>(x, emb, Xbf);
    lstm_persistent<<<NBLK, NTHR, 0, stream>>>(Xbf, W0b, W1b, b0, b1,
                                               G, C0, C1, H0, H1, H1f, bar);
    classifier<<<B_, 256, 0, stream>>>(H1f, V, bV, out);
}
